// DecoderOnlyTransformer_10239202034008
// MI455X (gfx1250) — matrix operand order repaired and checked by reading, not compiled, not run on hardware
//
#include <hip/hip_runtime.h>
#include <hip/hip_bf16.h>

// ---------------------------------------------------------------------------
// CDNA5 (gfx1250) decoder-only transformer forward.
// - all matmuls via v_wmma_f32_16x16x32_bf16 (wave32 WMMA)
// - global->LDS staging via GLOBAL_LOAD_ASYNC_TO_LDS_B128 (ASYNCcnt),
//   double-buffered so DMA overlaps WMMA compute
// - attention score strip (32 x 2048 f32) resident in the 320KB WGP LDS
// ---------------------------------------------------------------------------

typedef __bf16 bf16;
typedef __attribute__((ext_vector_type(16))) __bf16 v16bf;
typedef __attribute__((ext_vector_type(8)))  __bf16 v8bf;
typedef __attribute__((ext_vector_type(8)))  float  v8f;

// 16-bit A-matrix 16x32 layout (contiguous K order, matching B frags): element j
// (0..15) of the lane's fragment maps to K = 16*(lane>>4) + j; row M = lane & 15.
__device__ __forceinline__ int afrag_kidx(int lane, int j) {
  return ((lane >> 4) << 4) + j;
}

__device__ __forceinline__ v8f wmma_bf16(const v16bf& a, const v16bf& b, const v8f& c) {
  return __builtin_amdgcn_wmma_f32_16x16x32_bf16(false, a, false, b, (short)0, c, false, false);
}

// Async DMA: copy 16 bytes global -> LDS per lane (tracked by ASYNCcnt).
__device__ __forceinline__ void async_load_b128(const void* gaddr, void* lds) {
  asm volatile("global_load_async_to_lds_b128 %0, %1, off"
               :: "v"((unsigned)(unsigned long long)lds),
                  "v"((unsigned long long)gaddr)
               : "memory");
}
__device__ __forceinline__ void wait_async0() {
  asm volatile("s_wait_asynccnt 0" ::: "memory");
}

// ---------------------------------------------------------------------------
// fp32 -> bf16 conversion (weights)
// ---------------------------------------------------------------------------
__global__ void k_cvt(const float* in, bf16* out, size_t n) {
  size_t i = (size_t)blockIdx.x * blockDim.x + threadIdx.x;
  size_t stride = (size_t)gridDim.x * blockDim.x;
  for (; i < n; i += stride) out[i] = (bf16)in[i];
}

// ---------------------------------------------------------------------------
// Embedding gather + sinusoidal positional encoding (computed on the fly)
// ---------------------------------------------------------------------------
__global__ void k_embed(const int* tokens, const float* embed, float* x) {
  const int sIdx = blockIdx.x;            // 0 .. B*S-1
  const int s = sIdx & 2047;              // S = 2048
  const int tok = tokens[sIdx];
  const float* er = embed + (size_t)tok * 512;
  float* xr = x + (size_t)sIdx * 512;
  for (int c = threadIdx.x; c < 512; c += blockDim.x) {
    float di  = (float)(c & ~1);
    float dv  = __expf(di * (-9.210340371976184f / 512.0f));  // ln(1e4)
    float ang = (float)s * dv;
    float pe  = (c & 1) ? __cosf(ang) : __sinf(ang);
    xr[c] = er[c] + pe;
  }
}

// ---------------------------------------------------------------------------
// LayerNorm: one row (D=512) per wave; output bf16 for WMMA consumption
// ---------------------------------------------------------------------------
__global__ void k_ln(const float* x, const float* g, const float* b, bf16* out) {
  const int w = threadIdx.x >> 5, lane = threadIdx.x & 31;
  const int row = blockIdx.x * 8 + w;
  const float* xr = x + (size_t)row * 512;
  float s = 0.f, s2 = 0.f;
#pragma unroll
  for (int i = 0; i < 16; ++i) { float v = xr[lane + i * 32]; s += v; s2 += v * v; }
#pragma unroll
  for (int off = 16; off; off >>= 1) { s += __shfl_xor(s, off); s2 += __shfl_xor(s2, off); }
  const float mean = s * (1.f / 512.f);
  const float var  = s2 * (1.f / 512.f) - mean * mean;
  const float inv  = rsqrtf(var + 1e-5f);
  bf16* outr = out + (size_t)row * 512;
#pragma unroll
  for (int i = 0; i < 16; ++i) {
    int c = lane + i * 32;
    outr[c] = (bf16)((xr[c] - mean) * inv * g[c] + b[c]);
  }
}

// ---------------------------------------------------------------------------
// Tiled WMMA GEMM: C[M,N] = A[M,K](bf16) * B[K,N](bf16) [+bias][relu][+resid]
// Block tile 64x128, 8 waves as 2(M)x4(N), each wave 2x2 of 16x16 WMMA tiles.
// Double-buffered: A async-DMA'd to LDS (ASYNCcnt), B pipelined via registers
// one K-step ahead and scatter-stored transposed (contiguous B-frag reads).
// Ping-pong LDS buffers selected by offset arithmetic (no pointer-array
// initializers from LDS symbols -> avoids addrspacecast static-init lowering).
// ---------------------------------------------------------------------------
template <bool BIAS, bool RELU, bool RESID, bool OUTF, bool OUTB, bool NG>
__global__ void k_gemm(const bf16* A, const bf16* B, const float* bias,
                       const float* resid, float* Cf, bf16* Cb,
                       int M, int N, int K) {
  extern __shared__ char smem[];
  // buffer b: As at smem + b*12288 (64x32), Bt at smem + 4096 + b*12288 (128x32)
  const int tid = threadIdx.x, lane = tid & 31, w = tid >> 5;
  const int wm = w & 1, wn = w >> 1;
  const int mBase = blockIdx.y * 64;
  const int nBase = blockIdx.x * 128;

  v8f acc[2][2] = {};

  const int ar = tid >> 2, ac = (tid & 3) * 8;   // A staging coords (row, 8-col chunk)
  const int br = tid >> 3, bc = (tid & 7) * 16;  // B staging coords (k-row, 16-n chunk)
  const bool bok = !NG || (nBase + bc) < N;      // N % 16 == 0 -> chunk-uniform

  auto issueA = [&](int b, int k0) {
    async_load_b128(A + (size_t)(mBase + ar) * K + k0 + ac,
                    smem + b * 12288 + (ar * 32 + ac) * 2);
  };
  auto loadB = [&](int k0, v8bf& r0, v8bf& r1) {
    const bf16* g = B + (size_t)(k0 + br) * N + nBase + bc;
    if (bok) { r0 = *(const v8bf*)g; r1 = *(const v8bf*)(g + 8); }
    else     { r0 = (v8bf){};        r1 = (v8bf){}; }
  };

  v8bf brg0, brg1;
  issueA(0, 0);
  loadB(0, brg0, brg1);

  int buf = 0;
  for (int k0 = 0; k0 < K; k0 += 32) {
    wait_async0();                       // A tile for `buf` landed in LDS
    {                                    // scatter-store B regs transposed
      bf16* bt = (bf16*)(smem + 4096 + buf * 12288);
#pragma unroll
      for (int i = 0; i < 8; ++i) bt[(bc + i) * 32 + br] = brg0[i];
#pragma unroll
      for (int i = 0; i < 8; ++i) bt[(bc + 8 + i) * 32 + br] = brg1[i];
    }
    __syncthreads();                     // tile `buf` fully staged, prior readers done
    if (k0 + 32 < K) {                   // start next tile while we compute
      issueA(buf ^ 1, k0 + 32);
      loadB(k0 + 32, brg0, brg1);
    }

    const bf16* As = (const bf16*)(smem + buf * 12288);
    const bf16* Bt = (const bf16*)(smem + 4096 + buf * 12288);
    v16bf af[2], bfr[2];
#pragma unroll
    for (int mi = 0; mi < 2; ++mi) {
      const int row = wm * 32 + mi * 16 + (lane & 15);
#pragma unroll
      for (int j = 0; j < 16; ++j) af[mi][j] = As[row * 32 + afrag_kidx(lane, j)];
    }
    const int kg = (lane >> 4) << 4;     // K 0..15 vs 16..31 per lane half
#pragma unroll
    for (int ni = 0; ni < 2; ++ni) {
      const int col = wn * 32 + ni * 16 + (lane & 15);
      const bf16* bp = Bt + col * 32 + kg;
#pragma unroll
      for (int j = 0; j < 16; ++j) bfr[ni][j] = bp[j];
    }
#pragma unroll
    for (int mi = 0; mi < 2; ++mi)
#pragma unroll
      for (int ni = 0; ni < 2; ++ni)
        acc[mi][ni] = wmma_bf16(af[mi], bfr[ni], acc[mi][ni]);
    buf ^= 1;
  }

  // Epilogue. C/D layout: VGPR r -> M = r + 8*(lane>=16), N = lane & 15.
  const int m_off = (lane >> 4) << 3;
#pragma unroll
  for (int mi = 0; mi < 2; ++mi) {
#pragma unroll
    for (int ni = 0; ni < 2; ++ni) {
      const int n = nBase + wn * 32 + ni * 16 + (lane & 15);
      if (!NG || n < N) {
        const float bv = BIAS ? bias[n] : 0.f;
#pragma unroll
        for (int r = 0; r < 8; ++r) {
          const int m = mBase + wm * 32 + mi * 16 + m_off + r;
          float v = acc[mi][ni][r] + bv;
          if (RELU) v = fmaxf(v, 0.f);
          if (RESID) v += resid[(size_t)m * N + n];
          if (OUTF) Cf[(size_t)m * N + n] = v;
          if (OUTB) Cb[(size_t)m * N + n] = (bf16)v;
        }
      }
    }
  }
}

// ---------------------------------------------------------------------------
// Causal attention for one (b, h, 32-query block).
// Score strip [32 x 2048] f32 lives entirely in LDS (256KB of 320KB WGP LDS).
// Phase 1: S = (Q Kt) * inv_scale with causal mask  (WMMA, contraction HD=64)
// Phase 2: row softmax (wave-per-4-rows, shuffle reductions)
// Phase 3: O = P V (WMMA), V chunks pipelined via registers, double-buffered
// ---------------------------------------------------------------------------
__global__ void k_attn(const bf16* q, const bf16* k, const bf16* v, bf16* o) {
  extern __shared__ char smem[];
  float* Sc = (float*)smem;                         // 32 x 2048 scores
  bf16* Qs  = (bf16*)(smem + 262144);               // 32 x 64
  // V transpose buffers: smem + 266240 + vbuf*4096, each 64(d) x 32(key)
  const int tid = threadIdx.x, lane = tid & 31, w = tid >> 5;
  const int bx = blockIdx.x, hh = blockIdx.y, bb = blockIdx.z;
  const int qBase = bx * 32;
  const bf16* Qg = q + (size_t)bb * 2048 * 512 + hh * 64;
  const bf16* Kg = k + (size_t)bb * 2048 * 512 + hh * 64;
  const bf16* Vg = v + (size_t)bb * 2048 * 512 + hh * 64;
  bf16* Og       = o + (size_t)bb * 2048 * 512 + hh * 64;

  // stage Q block (32 x 64) via async DMA
  const int sr_ = tid >> 3, scb = (tid & 7) * 8;
  async_load_b128(Qg + (size_t)(qBase + sr_) * 512 + scb, Qs + sr_ * 64 + scb);
  wait_async0();
  __syncthreads();

  const float inv_scale = 0.044194173824159216f;  // 1/sqrt(D=512), per reference
  const int mi = w & 1;
  const int ntMax = 2 * (bx + 1);                 // key tiles with any live key
  for (int nt = (w >> 1); nt < 128; nt += 4) {
    const int key = nt * 16 + (lane & 15);
    const int lrowBase = mi * 16 + ((lane >> 4) << 3);
    if (nt >= ntMax) {                            // fully masked tile
#pragma unroll
      for (int r = 0; r < 8; ++r) Sc[(lrowBase + r) * 2048 + key] = -1e30f;
      continue;
    }
    v16bf a0, a1, b0, b1;
    {
      const int kg = (lane >> 4) << 4;
      const bf16* kr = Kg + (size_t)key * 512;    // B frag: per-lane key row
#pragma unroll
      for (int j = 0; j < 16; ++j) { b0[j] = kr[kg + j]; b1[j] = kr[32 + kg + j]; }
      const int row = mi * 16 + (lane & 15);
#pragma unroll
      for (int j = 0; j < 16; ++j) {
        const int kk = afrag_kidx(lane, j);
        a0[j] = Qs[row * 64 + kk];
        a1[j] = Qs[row * 64 + 32 + kk];
      }
    }
    v8f acc = {};
    acc = wmma_bf16(a0, b0, acc);
    acc = wmma_bf16(a1, b1, acc);
#pragma unroll
    for (int r = 0; r < 8; ++r) {
      const int lrow = lrowBase + r;
      float val = acc[r] * inv_scale;
      if (key > qBase + lrow) val = -1e30f;       // causal mask
      Sc[lrow * 2048 + key] = val;
    }
  }
  __syncthreads();

  // prefetch first V chunk into registers (hides latency under softmax)
  const int kl = tid >> 3, db = (tid & 7) * 8;
  v8bf vreg = *(const v8bf*)(Vg + (size_t)kl * 512 + db);

  // softmax: wave w handles rows 4w..4w+3, lanes stride the 2048 columns
  for (int rr = 0; rr < 4; ++rr) {
    float* sr = Sc + (w * 4 + rr) * 2048;
    float m = -3.0e38f;
    for (int c = lane; c < 2048; c += 32) m = fmaxf(m, sr[c]);
#pragma unroll
    for (int off = 16; off; off >>= 1) m = fmaxf(m, __shfl_xor(m, off));
    float sum = 0.f;
    for (int c = lane; c < 2048; c += 32) { float e = __expf(sr[c] - m); sr[c] = e; sum += e; }
#pragma unroll
    for (int off = 16; off; off >>= 1) sum += __shfl_xor(sum, off);
    const float inv = 1.f / sum;
    for (int c = lane; c < 2048; c += 32) sr[c] *= inv;
  }

  // P @ V : waves as 2(M) x 4(N=HD tiles); only key chunks <= diagonal live
  const int pmi = w >> 2, ni = w & 3;
  v8f acc = {};
  int vbuf = 0;
  for (int kc = 0; kc <= bx; ++kc) {
    {  // scatter-store V regs transposed into buffer `vbuf`
      bf16* vt = (bf16*)(smem + 266240 + vbuf * 4096);
#pragma unroll
      for (int i = 0; i < 8; ++i) vt[(db + i) * 32 + kl] = vreg[i];
    }
    __syncthreads();
    if (kc < bx)  // pipeline next chunk
      vreg = *(const v8bf*)(Vg + (size_t)((kc + 1) * 32 + kl) * 512 + db);
    v16bf a, bfr;
    {
      const int row = pmi * 16 + (lane & 15);
      const float* sr = Sc + row * 2048 + kc * 32;
#pragma unroll
      for (int j = 0; j < 16; ++j) a[j] = (bf16)sr[afrag_kidx(lane, j)];
      const int d = ni * 16 + (lane & 15);
      const int kg = (lane >> 4) << 4;
      const bf16* vt = (const bf16*)(smem + 266240 + vbuf * 4096) + d * 32 + kg;
#pragma unroll
      for (int j = 0; j < 16; ++j) bfr[j] = vt[j];
    }
    acc = wmma_bf16(a, bfr, acc);
    vbuf ^= 1;
  }
  {
    const int m_off = (lane >> 4) << 3;
    const int n = ni * 16 + (lane & 15);
#pragma unroll
    for (int r = 0; r < 8; ++r) {
      const int qi = qBase + pmi * 16 + m_off + r;
      Og[(size_t)qi * 512 + n] = (bf16)acc[r];
    }
  }
}

// ---------------------------------------------------------------------------
// Host orchestration
// ---------------------------------------------------------------------------
extern "C" void kernel_launch(void* const* d_in, const int* in_sizes, int n_in,
                              void* d_out, int out_size, void* d_ws, size_t ws_size,
                              hipStream_t stream) {
  (void)in_sizes; (void)n_in; (void)out_size; (void)ws_size;
  const int D = 512, L = 6, FF = 2048, V = 8000, M = 4 * 2048;

  const int*   tokens = (const int*)d_in[0];
  const float* embed  = (const float*)d_in[1];
  const float* wq     = (const float*)d_in[2];
  const float* wk     = (const float*)d_in[3];
  const float* wv     = (const float*)d_in[4];
  const float* wo     = (const float*)d_in[5];
  const float* bo     = (const float*)d_in[6];
  const float* w1     = (const float*)d_in[7];
  const float* b1     = (const float*)d_in[8];
  const float* w2     = (const float*)d_in[9];
  const float* b2     = (const float*)d_in[10];
  const float* ln_g   = (const float*)d_in[11];
  const float* ln_b   = (const float*)d_in[12];
  const float* lnf_g  = (const float*)d_in[13];
  const float* lnf_b  = (const float*)d_in[14];
  const float* w_out  = (const float*)d_in[15];
  const float* b_out  = (const float*)d_in[16];
  float* out = (float*)d_out;

  size_t off = 0;
  auto arena = [&](size_t bytes) -> char* {
    char* p = (char*)d_ws + off;
    off += (bytes + 255) & ~(size_t)255;
    return p;
  };
  bf16* wqB  = (bf16*)arena((size_t)L * D * D * 2);
  bf16* wkB  = (bf16*)arena((size_t)L * D * D * 2);
  bf16* wvB  = (bf16*)arena((size_t)L * D * D * 2);
  bf16* woB  = (bf16*)arena((size_t)L * D * D * 2);
  bf16* w1B  = (bf16*)arena((size_t)L * D * FF * 2);
  bf16* w2B  = (bf16*)arena((size_t)L * FF * D * 2);
  bf16* woutB= (bf16*)arena((size_t)D * V * 2);
  float* x   = (float*)arena((size_t)M * D * 4);
  bf16* h    = (bf16*)arena((size_t)M * D * 2);
  bf16* qb   = (bf16*)arena((size_t)M * D * 2);
  bf16* kb   = (bf16*)arena((size_t)M * D * 2);
  bf16* vb   = (bf16*)arena((size_t)M * D * 2);
  bf16* ao   = (bf16*)arena((size_t)M * D * 2);
  bf16* ff   = (bf16*)arena((size_t)M * FF * 2);

  // weight conversion fp32 -> bf16 (bf16 working set ~46 MB: L2-resident)
  k_cvt<<<2048, 256, 0, stream>>>(wq, wqB, (size_t)L * D * D);
  k_cvt<<<2048, 256, 0, stream>>>(wk, wkB, (size_t)L * D * D);
  k_cvt<<<2048, 256, 0, stream>>>(wv, wvB, (size_t)L * D * D);
  k_cvt<<<2048, 256, 0, stream>>>(wo, woB, (size_t)L * D * D);
  k_cvt<<<4096, 256, 0, stream>>>(w1, w1B, (size_t)L * D * FF);
  k_cvt<<<4096, 256, 0, stream>>>(w2, w2B, (size_t)L * FF * D);
  k_cvt<<<4096, 256, 0, stream>>>(w_out, woutB, (size_t)D * V);

  k_embed<<<M, 256, 0, stream>>>(tokens, embed, x);

  const size_t gemmLds = 2 * (64 * 32 * 2 + 128 * 32 * 2);      // 24 KB (double-buffered)
  const size_t attnLds = 262144 + 4096 + 2 * 4096;              // 268 KB of 320 KB WGP LDS
  const dim3 gProj(4, 128), gFF1(16, 128), gLogits(63, 128), gAttn(64, 8, 4);

  for (int l = 0; l < L; ++l) {
    k_ln<<<1024, 256, 0, stream>>>(x, ln_g + l * D, ln_b + l * D, h);
    k_gemm<false,false,false,false,true,false><<<gProj, 256, gemmLds, stream>>>(
        h, wqB + (size_t)l * D * D, nullptr, nullptr, nullptr, qb, M, D, D);
    k_gemm<false,false,false,false,true,false><<<gProj, 256, gemmLds, stream>>>(
        h, wkB + (size_t)l * D * D, nullptr, nullptr, nullptr, kb, M, D, D);
    k_gemm<false,false,false,false,true,false><<<gProj, 256, gemmLds, stream>>>(
        h, wvB + (size_t)l * D * D, nullptr, nullptr, nullptr, vb, M, D, D);
    k_attn<<<gAttn, 256, attnLds, stream>>>(qb, kb, vb, ao);
    k_gemm<true,false,true,true,false,false><<<gProj, 256, gemmLds, stream>>>(
        ao, woB + (size_t)l * D * D, bo + l * D, x, x, nullptr, M, D, D);
    k_ln<<<1024, 256, 0, stream>>>(x, ln_g + l * D, ln_b + l * D, h);
    k_gemm<true,true,false,false,true,false><<<gFF1, 256, gemmLds, stream>>>(
        h, w1B + (size_t)l * D * FF, b1 + l * FF, nullptr, nullptr, ff, M, FF, D);
    k_gemm<true,false,true,true,false,false><<<gProj, 256, gemmLds, stream>>>(
        ff, w2B + (size_t)l * FF * D, b2 + l * D, x, x, nullptr, M, D, FF);
  }

  k_ln<<<1024, 256, 0, stream>>>(x, lnf_g, lnf_b, h);
  k_gemm<true,false,false,true,false,true><<<gLogits, 256, gemmLds, stream>>>(
      h, woutB, b_out, nullptr, out, nullptr, M, V, D);
}
